// MambaModel_15393162789503
// MI455X (gfx1250) — matrix operand order repaired and checked by reading, not compiled, not run on hardware
//
#include <hip/hip_runtime.h>
#include <hip/hip_bf16.h>
#include <cmath>

// Problem constants (fixed by the reference).
#define L_SEQ   2048
#define DMODEL  1024
#define DINNER  2048
#define DSTATE  16
#define DTRANK  64
#define DCONV   4
#define BATCH   8
#define XPROJ_F (DTRANK + 2*DSTATE)   // 96

typedef __attribute__((ext_vector_type(16))) _Float16     v16h;
typedef __attribute__((ext_vector_type(8)))  _Float16     v8h;
typedef __attribute__((ext_vector_type(8)))  float        v8f;
typedef __attribute__((ext_vector_type(4)))  unsigned int v4u;
typedef __attribute__((ext_vector_type(8)))  unsigned int v8u;

// ---------------------------------------------------------------- utilities
__global__ __launch_bounds__(256) void k_cvt_f16(const float* __restrict__ s,
                                                 _Float16* __restrict__ d, int n) {
  int i = blockIdx.x * 256 + threadIdx.x;
  if (i < n) d[i] = (_Float16)s[i];
}

__global__ __launch_bounds__(256) void k_aneg(const float* __restrict__ alog,
                                              float* __restrict__ an, int n) {
  int i = blockIdx.x * 256 + threadIdx.x;
  if (i < n) an[i] = -__expf(alog[i]);
}

// --------------------------------------------------------- TDM tile loader
// Issues TENSOR_LOAD_TO_LDS for a 2D tile: `rows` rows of 32 f16 elements,
// row stride `k_elems` (elements), from byte address gaddr into LDS at
// lds_off. D# layout per CDNA5 ISA ch.8 (group0/1; groups 2-3 zero = 2D).
__device__ __forceinline__ void tdm_load_tile32(unsigned lds_off,
                                                unsigned long long gaddr,
                                                unsigned k_elems,
                                                unsigned rows) {
  v4u g0, gz;
  v8u g1;
  g0[0] = 1u;                                            // count=1, user mode
  g0[1] = lds_off;                                       // lds_addr
  g0[2] = (unsigned)gaddr;                               // global_addr[31:0]
  g0[3] = ((unsigned)(gaddr >> 32) & 0x01FFFFFFu)        // global_addr[56:32]
          | 0x80000000u;                                 // type=2 (image)
  g1[0] = 1u << 16;                                      // data_size=2 bytes
  g1[1] = (k_elems & 0xFFFFu) << 16;                     // tensor_dim0 lo16
  g1[2] = (k_elems >> 16) & 0xFFFFu;                     // tensor_dim0 hi16 | td1 lo16(=0)
  g1[3] = 0x10u | (32u << 16);                           // td1 hi16 (1M) | tile_dim0=32
  g1[4] = rows;                                          // tile_dim1 | tile_dim2=0
  g1[5] = k_elems;                                       // tensor_dim0_stride lo32
  g1[6] = 0u;
  g1[7] = 0u;
  gz[0] = 0u; gz[1] = 0u; gz[2] = 0u; gz[3] = 0u;        // groups 2/3: 2D tile
  asm volatile("tensor_load_to_lds %0, %1, %2, %3"
               :: "s"(g0), "s"(g1), "s"(gz), "s"(gz)
               : "memory");
}

// ------------------------------------------- TDM-staged WMMA GEMM: C = A·Wᵀ
// A: M×K f16, W: N×K f16 (row-major), C: M×N f32. M%128==0, N%64==0, K%32==0.
// Block = 8 waves: 128M×64N tile. Per 32-wide K step the TDM DMAs the A tile
// (128×32) and W tile (64×32) into LDS (double-buffered, TENSORcnt-tracked);
// waves consume fragments from LDS and run 4 WMMAs each.
__global__ __launch_bounds__(256)
void k_gemm_tdm(const _Float16* __restrict__ A, const _Float16* __restrict__ W,
                float* __restrict__ C, int M, int N, int K) {
  __shared__ _Float16 sA[2][128 * 32];   // 16 KB
  __shared__ _Float16 sW[2][64 * 32];    //  8 KB

  const int lane = threadIdx.x & 31;
  const int wave = threadIdx.x >> 5;
  const int half = lane >> 4;            // 0: lanes 0-15, 1: lanes 16-31
  const int l16  = lane & 15;
  const int ako  = half * 16;            // A: K groups {0-15}/{16-31}
  const int bko  = half * 16;            // B: K groups {0-15}/{16-31}
  const int m0   = blockIdx.y * 128;
  const int n0   = blockIdx.x * 64;

  const unsigned long long aBase =
      (unsigned long long)(const char*)(A + (size_t)m0 * K);
  const unsigned long long wBase =
      (unsigned long long)(const char*)(W + (size_t)n0 * K);

  v8f zero = {};
  v8f acc[4];
#pragma unroll
  for (int t = 0; t < 4; ++t) acc[t] = zero;

  // preload k-step 0 into buffer 0
  if (wave == 0) {
    tdm_load_tile32((unsigned)(size_t)&sA[0][0], aBase, (unsigned)K, 128u);
    tdm_load_tile32((unsigned)(size_t)&sW[0][0], wBase, (unsigned)K, 64u);
    __builtin_amdgcn_s_wait_tensorcnt(0);
  }
  __syncthreads();

  for (int k0 = 0; k0 < K; k0 += 32) {
    const int cur = (k0 >> 5) & 1;
    const int nxt = cur ^ 1;
    const bool more = (k0 + 32) < K;

    // kick off DMA for the next K slab while we compute on the current one
    if (wave == 0 && more) {
      const unsigned long long koff = (unsigned long long)(k0 + 32) * 2;
      tdm_load_tile32((unsigned)(size_t)&sA[nxt][0], aBase + koff, (unsigned)K, 128u);
      tdm_load_tile32((unsigned)(size_t)&sW[nxt][0], wBase + koff, (unsigned)K, 64u);
    }

    const _Float16* aRow = &sA[cur][(wave * 16 + l16) * 32];
    v8h alo = *(const v8h*)(aRow + ako);
    v8h ahi = *(const v8h*)(aRow + ako + 8);
    v16h a;
#pragma unroll
    for (int i = 0; i < 8; ++i) { a[i] = alo[i]; a[i + 8] = ahi[i]; }

    v16h b[4];
#pragma unroll
    for (int t = 0; t < 4; ++t)
      b[t] = *(const v16h*)(&sW[cur][(t * 16 + l16) * 32 + bko]);
#pragma unroll
    for (int t = 0; t < 4; ++t)
      acc[t] = __builtin_amdgcn_wmma_f32_16x16x32_f16(
          false, a, false, b[t], (short)0, acc[t], false, false);

    if (wave == 0 && more) __builtin_amdgcn_s_wait_tensorcnt(0);
    __syncthreads();   // nobody may still read `cur` when it is re-DMA'd
  }

#pragma unroll
  for (int t = 0; t < 4; ++t) {
    const int n = n0 + t * 16 + l16;
    float* Cp = C + (size_t)(m0 + wave * 16 + half * 8) * N + n;
#pragma unroll
    for (int r = 0; r < 8; ++r) Cp[(size_t)r * N] = acc[t][r];
  }
}

// ------------------------------------- direct-load WMMA GEMM (small shapes)
// Same math as above without LDS staging. N%(16*NT)==0, M%128==0, K%32==0.
template<int NT>
__global__ __launch_bounds__(256)
void k_gemm_xwT(const _Float16* __restrict__ A, const _Float16* __restrict__ W,
                float* __restrict__ C, int M, int N, int K) {
  const int lane = threadIdx.x & 31;
  const int wave = threadIdx.x >> 5;
  const int m0   = (blockIdx.y * 8 + wave) * 16;
  const int n0   = blockIdx.x * (16 * NT);
  const int half = lane >> 4;
  const int l16  = lane & 15;
  const int ako  = half * 16;
  const int bko  = half * 16;

  v8f zero = {};
  v8f acc[NT];
#pragma unroll
  for (int t = 0; t < NT; ++t) acc[t] = zero;

  const _Float16* Arow = A + (size_t)(m0 + l16) * K;

  for (int k0 = 0; k0 < K; k0 += 32) {
    if (k0 + 32 < K) __builtin_prefetch(Arow + k0 + 32 + ako, 0, 3);
    v8h alo = *(const v8h*)(Arow + k0 + ako);
    v8h ahi = *(const v8h*)(Arow + k0 + ako + 8);
    v16h a;
#pragma unroll
    for (int i = 0; i < 8; ++i) { a[i] = alo[i]; a[i + 8] = ahi[i]; }
    v16h b[NT];
#pragma unroll
    for (int t = 0; t < NT; ++t) {
      const int n = n0 + t * 16 + l16;
      b[t] = *(const v16h*)(W + (size_t)n * K + k0 + bko);
    }
#pragma unroll
    for (int t = 0; t < NT; ++t)
      acc[t] = __builtin_amdgcn_wmma_f32_16x16x32_f16(
          false, a, false, b[t], (short)0, acc[t], false, false);
  }

#pragma unroll
  for (int t = 0; t < NT; ++t) {
    const int n = n0 + t * 16 + l16;
    float* Cp = C + (size_t)(m0 + half * 8) * N + n;
#pragma unroll
    for (int r = 0; r < 8; ++r) Cp[(size_t)r * N] = acc[t][r];
  }
}

// ------------------------------------------- causal depthwise conv4 + SiLU
__global__ __launch_bounds__(256)
void k_conv(const float* __restrict__ xz, const float* __restrict__ cw,
            const float* __restrict__ cb, float* __restrict__ xc32,
            _Float16* __restrict__ xc16) {
  int i = blockIdx.x * 256 + threadIdx.x;         // over L*DINNER
  int l = i >> 11, e = i & (DINNER - 1);
  float s = cb[e];
#pragma unroll
  for (int k = 0; k < DCONV; ++k) {
    int l2 = l - (DCONV - 1) + k;
    if (l2 >= 0) s += cw[e * DCONV + k] * xz[(size_t)l2 * (2 * DINNER) + e];
  }
  s = s / (1.f + __expf(-s));                     // SiLU
  xc32[i] = s;
  xc16[i] = (_Float16)s;
}

__global__ __launch_bounds__(256)
void k_dtext(const float* __restrict__ xdbl, _Float16* __restrict__ dt16) {
  int i = blockIdx.x * 256 + threadIdx.x;         // over L*DTRANK
  int l = i >> 6, r = i & (DTRANK - 1);
  dt16[i] = (_Float16)xdbl[(size_t)l * XPROJ_F + r];
}

__global__ __launch_bounds__(256)
void k_softplus(float* __restrict__ dlt, const float* __restrict__ bias) {
  int i = blockIdx.x * 256 + threadIdx.x;         // over L*DINNER
  int e = i & (DINNER - 1);
  float v = dlt[i] + bias[e];
  dlt[i] = (v > 20.f) ? v : log1pf(__expf(v));
}

// -------------------------------------------------- selective scan over L
// One lane per (channel e, state n); y reduced across the 16-lane state
// group with shfl_xor. dy: delta in, y out (in place).
__global__ __launch_bounds__(256)
void k_scan(float* __restrict__ dy, const float* __restrict__ xc,
            const float* __restrict__ xdbl, const float* __restrict__ an) {
  int t = blockIdx.x * 256 + threadIdx.x;         // 0 .. DINNER*DSTATE-1
  int n = t & (DSTATE - 1);
  int e = t >> 4;
  const float A = an[e * DSTATE + n];
  float h = 0.f;
  for (int l = 0; l < L_SEQ; ++l) {
    float d  = dy[(size_t)l * DINNER + e];
    float xv = xc[(size_t)l * DINNER + e];
    float Bn = xdbl[(size_t)l * XPROJ_F + DTRANK + n];
    float Cn = xdbl[(size_t)l * XPROJ_F + DTRANK + DSTATE + n];
    h = __expf(d * A) * h + (d * xv) * Bn;
    float y = h * Cn;
    y += __shfl_xor(y, 1, 16);
    y += __shfl_xor(y, 2, 16);
    y += __shfl_xor(y, 4, 16);
    y += __shfl_xor(y, 8, 16);
    if (n == 0) dy[(size_t)l * DINNER + e] = y;
  }
}

__global__ __launch_bounds__(256)
void k_gate(const float* __restrict__ ys, const float* __restrict__ xc32,
            const float* __restrict__ xz, const float* __restrict__ Dv,
            _Float16* __restrict__ y16) {
  int i = blockIdx.x * 256 + threadIdx.x;         // over L*DINNER
  int l = i >> 11, e = i & (DINNER - 1);
  float y = ys[i] + xc32[i] * Dv[e];
  float z = xz[(size_t)l * (2 * DINNER) + DINNER + e];
  y *= z / (1.f + __expf(-z));
  y16[i] = (_Float16)y;
}

__global__ __launch_bounds__(256)
void k_bcast(const float* __restrict__ o1, float* __restrict__ out) {
  int i = blockIdx.x * 256 + threadIdx.x;         // over L*DMODEL
  float v = o1[i];
#pragma unroll
  for (int b = 0; b < BATCH; ++b)
    out[(size_t)b * L_SEQ * DMODEL + i] = v;
}

// ---------------------------------------------------------------- launcher
extern "C" void kernel_launch(void* const* d_in, const int* in_sizes, int n_in,
                              void* d_out, int out_size, void* d_ws, size_t ws_size,
                              hipStream_t stream) {
  const float* pe    = (const float*)d_in[0];
  const float* w_in  = (const float*)d_in[1];
  const float* cw    = (const float*)d_in[2];
  const float* cb    = (const float*)d_in[3];
  const float* w_x   = (const float*)d_in[4];
  const float* w_dt  = (const float*)d_in[5];
  const float* b_dt  = (const float*)d_in[6];
  const float* alog  = (const float*)d_in[7];
  const float* Dv    = (const float*)d_in[8];
  const float* w_out = (const float*)d_in[9];
  float* out = (float*)d_out;

  // d_out doubles as scratch for the three largest f32 intermediates
  // (32 + 16 + 16 MB = 64 MB = out_size); k_bcast overwrites all of it last.
  float* xz    = out;                               // (L, 4096)
  float* xc32  = out + (size_t)L_SEQ * (2 * DINNER);// (L, 2048)
  float* dlt   = xc32 + (size_t)L_SEQ * DINNER;     // (L, 2048) delta -> y

  char* ws = (char*)d_ws;
  size_t off = 0;
  auto take = [&](size_t bytes) -> char* {
    char* p = ws + off;
    off = (off + bytes + 255) & ~(size_t)255;
    return p;
  };
  _Float16* w1h  = (_Float16*)take((size_t)2 * DINNER * DMODEL * 2);   // 4096x1024
  _Float16* w2h  = (_Float16*)take((size_t)XPROJ_F * DINNER * 2);      // 96x2048
  _Float16* w3h  = (_Float16*)take((size_t)DINNER * DTRANK * 2);       // 2048x64
  _Float16* w4h  = (_Float16*)take((size_t)DMODEL * DINNER * 2);       // 1024x2048
  _Float16* peh  = (_Float16*)take((size_t)L_SEQ * DMODEL * 2);        // 2048x1024
  _Float16* xc16 = (_Float16*)take((size_t)L_SEQ * DINNER * 2);        // also y16
  float*    xdbl = (float*)   take((size_t)L_SEQ * XPROJ_F * 4);       // 2048x96
  _Float16* dt16 = (_Float16*)take((size_t)L_SEQ * DTRANK * 2);        // 2048x64
  float*    aneg = (float*)   take((size_t)DINNER * DSTATE * 4);       // 2048x16
  float*    out1 = (float*)   take((size_t)L_SEQ * DMODEL * 4);        // 2048x1024
  (void)ws_size; (void)in_sizes; (void)n_in; (void)out_size;

  auto blk = [](int n) { return (n + 255) / 256; };

  // 1) precision conversions + A = -exp(A_log)
  k_cvt_f16<<<blk(2*DINNER*DMODEL), 256, 0, stream>>>(w_in,  w1h, 2*DINNER*DMODEL);
  k_cvt_f16<<<blk(XPROJ_F*DINNER),  256, 0, stream>>>(w_x,   w2h, XPROJ_F*DINNER);
  k_cvt_f16<<<blk(DINNER*DTRANK),   256, 0, stream>>>(w_dt,  w3h, DINNER*DTRANK);
  k_cvt_f16<<<blk(DMODEL*DINNER),   256, 0, stream>>>(w_out, w4h, DMODEL*DINNER);
  k_cvt_f16<<<blk(L_SEQ*DMODEL),    256, 0, stream>>>(pe,    peh, L_SEQ*DMODEL);
  k_aneg   <<<blk(DINNER*DSTATE),   256, 0, stream>>>(alog,  aneg, DINNER*DSTATE);

  // 2) GEMM1 (TDM-staged): xz = pe @ in_proj_wᵀ   (2048x1024x4096)
  k_gemm_tdm<<<dim3((2*DINNER)/64, L_SEQ/128), 256, 0, stream>>>(
      peh, w1h, xz, L_SEQ, 2*DINNER, DMODEL);

  // 3) causal depthwise conv + SiLU
  k_conv<<<blk(L_SEQ*DINNER), 256, 0, stream>>>(xz, cw, cb, xc32, xc16);

  // 4) GEMM2: x_dbl = xconv @ x_proj_wᵀ  (2048x2048x96)
  k_gemm_xwT<2><<<dim3(XPROJ_F/32, L_SEQ/128), 256, 0, stream>>>(
      xc16, w2h, xdbl, L_SEQ, XPROJ_F, DINNER);

  // 5) GEMM3: delta_pre = dt @ dt_proj_wᵀ  (2048x64x2048)
  k_dtext<<<blk(L_SEQ*DTRANK), 256, 0, stream>>>(xdbl, dt16);
  k_gemm_xwT<4><<<dim3(DINNER/64, L_SEQ/128), 256, 0, stream>>>(
      dt16, w3h, dlt, L_SEQ, DINNER, DTRANK);
  k_softplus<<<blk(L_SEQ*DINNER), 256, 0, stream>>>(dlt, b_dt);

  // 6) selective scan (lane per (e,n), shfl reduction over state dim)
  k_scan<<<(DINNER * DSTATE) / 256, 256, 0, stream>>>(dlt, xc32, xdbl, aneg);

  // 7) skip + gating -> f16
  k_gate<<<blk(L_SEQ*DINNER), 256, 0, stream>>>(dlt, xc32, xz, Dv, xc16);

  // 8) GEMM4 (TDM-staged): out1 = y @ out_proj_wᵀ  (2048x2048x1024)
  k_gemm_tdm<<<dim3(DMODEL/64, L_SEQ/128), 256, 0, stream>>>(
      xc16, w4h, out1, L_SEQ, DMODEL, DINNER);

  // 9) replicate over the (identical) batch dimension
  k_bcast<<<blk(L_SEQ*DMODEL), 256, 0, stream>>>(out1, out);
}
